// GAT_62182536511748
// MI455X (gfx1250) — matrix operand order repaired and checked by reading, not compiled, not run on hardware
//
#include <hip/hip_runtime.h>

#define NNODES   50000
#define DEGREE   16
#define IN_F     256
#define HH_F     128
#define OUT_F    40
#define OUT_PAD  48
#define NEG_SLOPE 0.2f

typedef _Float16 h16;
typedef __attribute__((ext_vector_type(8)))  _Float16 v8h;
typedef __attribute__((ext_vector_type(16))) _Float16 v16h;
typedef __attribute__((ext_vector_type(8)))  float    v8f;

// ---------------------------------------------------------------- converts
__global__ void cvt_f32_to_f16_kernel(const float* __restrict__ in,
                                      h16* __restrict__ out, int n) {
  int i = blockIdx.x * blockDim.x + threadIdx.x;
  if (i < n) out[i] = (h16)in[i];
}

// Wt[n][k] = W[k][n] as f16; columns >= Ncols zero-padded (for OUT_F->48 pad).
__global__ void transpose_f16_kernel(const float* __restrict__ W,
                                     h16* __restrict__ Wt,
                                     int K, int Ncols, int Npad) {
  int i = blockIdx.x * blockDim.x + threadIdx.x;
  if (i >= K * Npad) return;
  int n = i / K, k = i - n * K;
  Wt[(size_t)n * K + k] = (n < Ncols) ? (h16)W[(size_t)k * Ncols + n] : (h16)0.f;
}

// ---------------------------------------------------------------- WMMA GEMM
// C[M x NT*16] = A[M x K] (f16 row-major) * Bt^T  (Bt is [NT*16 x K] f16).
// One wave per 16-row panel; the wave owns ALL NT column tiles so the A
// fragment is loaded once per k-step and reused by NT back-to-back WMMAs.
// M multiple of 16 (50000 = 3125*16), K multiple of 32 (compile-time).
template<int K, int NT>
__global__ __launch_bounds__(32)
void wmma_gemm_kernel(const h16* __restrict__ A, const h16* __restrict__ Bt,
                      float* __restrict__ C) {
  const int lane = threadIdx.x;        // 0..31, EXEC all ones (no divergence)
  const int n    = lane & 15;
  const int hi   = lane >> 4;          // lane half selector
  const int rowBase = blockIdx.x << 4;
  constexpr int Nc = NT * 16;

  // A 16x32 layout: lanes 0-15 hold K 0..15, lanes 16-31 hold K 16..31
  const int kbA = hi << 4;             // 0 or 16
  // B 32x16 layout, same contiguous k order as A: lanes 0-15 hold K 0..15, lanes 16-31 hold K 16..31
  const int kbB = hi << 4;             // 0 or 16

  const h16* Arow  = A  + (size_t)(rowBase + n) * K;
  const h16* Bcol0 = Bt + (size_t)n * K;

  v8f c[NT];
#pragma unroll
  for (int t = 0; t < NT; ++t) c[t] = (v8f){};

#pragma unroll
  for (int k0 = 0; k0 < K; k0 += 32) {
    v16h a  = *(const v16h*)(Arow + k0 + kbA);
#pragma unroll
    for (int t = 0; t < NT; ++t) {
      v16h b = *(const v16h*)(Bcol0 + (size_t)t * 16 * K + k0 + kbB);
      c[t] = __builtin_amdgcn_wmma_f32_16x16x32_f16(false, a, false, b,
                                                    (short)0, c[t], false, false);
    }
  }

  // C/D layout: VGPR r -> row rowBase + r + hi*8, col t*16 + n
  float* Crow = C + (size_t)(rowBase + (hi << 3)) * Nc + n;
#pragma unroll
  for (int t = 0; t < NT; ++t)
#pragma unroll
    for (int r = 0; r < 8; ++r)
      Crow[(size_t)r * Nc + t * 16] = c[t][r];
}

// ------------------------------------------------- attention scores el / er
// el[n,h] = dot(feat[n, h*D : (h+1)*D], al[h]);  er likewise.
__global__ void attn_scores_kernel(const float* __restrict__ feat,
                                   const float* __restrict__ al,
                                   const float* __restrict__ ar,
                                   float* __restrict__ el,
                                   float* __restrict__ er,
                                   int H, int D, int ldf) {
  int i = blockIdx.x * blockDim.x + threadIdx.x;
  if (i >= NNODES * H) return;
  int node = i / H, h = i - node * H;
  const float* f = feat + (size_t)node * ldf + h * D;
  const float* a = al + h * D;
  const float* r = ar + h * D;
  float sl = 0.f, sr = 0.f;
  for (int d = 0; d < D; ++d) {
    sl = fmaf(f[d], a[d], sl);
    sr = fmaf(f[d], r[d], sr);
  }
  el[i] = sl;
  er[i] = sr;
}

// ---------------------------------------- per-node softmax + gather-aggregate
// dst = repeat(arange(N), DEG): node n owns edges [n*DEG, n*DEG+DEG).
// One block per node: LDS-staged leaky-relu + softmax over 16 edges, then
// weighted gather of feat[src] rows, fused bias (+ReLU) and f16 requantize.
template<int H, int D, int LDF, bool RELU>
__global__ void gat_aggregate_kernel(const float* __restrict__ feat,  // [N, LDF]
                                     const float* __restrict__ el,    // [N, H]
                                     const float* __restrict__ er,    // [N, H]
                                     const int*   __restrict__ src,   // [N*DEG]
                                     const float* __restrict__ bias,  // [H*D]
                                     float* __restrict__ outF,        // [N, H*D] or null
                                     h16*   __restrict__ outH) {      // [N, H*D] or null
  const int node = blockIdx.x;
  __shared__ int   s_src[DEGREE];
  __shared__ float s_a[DEGREE * H];
  const int t = threadIdx.x;

  if (t < DEGREE) s_src[t] = src[node * DEGREE + t];
  __syncthreads();

  if (t < DEGREE * H) {
    const int j = t / H, h = t - (t / H) * H;
    float e = el[(size_t)s_src[j] * H + h] + er[(size_t)node * H + h];
    s_a[j * H + h] = (e > 0.f) ? e : NEG_SLOPE * e;
  }
  __syncthreads();

  if (t < H) {
    float m = -3.0e38f;
#pragma unroll
    for (int j = 0; j < DEGREE; ++j) m = fmaxf(m, s_a[j * H + t]);
    float s = 0.f;
#pragma unroll
    for (int j = 0; j < DEGREE; ++j) {
      float ex = __expf(s_a[j * H + t] - m);
      s_a[j * H + t] = ex;
      s += ex;
    }
    const float inv = 1.f / s;
#pragma unroll
    for (int j = 0; j < DEGREE; ++j) s_a[j * H + t] *= inv;
  }
  __syncthreads();

  if (t < H * D) {
    const int h = t / D;
    float acc = 0.f;
#pragma unroll
    for (int j = 0; j < DEGREE; ++j)
      acc = fmaf(s_a[j * H + h], feat[(size_t)s_src[j] * LDF + t], acc);
    acc += bias[t];
    if (RELU) acc = fmaxf(acc, 0.f);
    if (outF) outF[(size_t)node * (H * D) + t] = acc;
    if (outH) outH[(size_t)node * (H * D) + t] = (h16)acc;
  }
}

// ---------------------------------------------------------------- log_softmax
__global__ void log_softmax_kernel(const float* __restrict__ logits,
                                   float* __restrict__ out) {
  int n = blockIdx.x * blockDim.x + threadIdx.x;
  if (n >= NNODES) return;
  const float* x = logits + (size_t)n * OUT_F;
  float m = x[0];
  for (int c = 1; c < OUT_F; ++c) m = fmaxf(m, x[c]);
  float s = 0.f;
  for (int c = 0; c < OUT_F; ++c) s += __expf(x[c] - m);
  const float lse = m + __logf(s);
  float* o = out + (size_t)n * OUT_F;
  for (int c = 0; c < OUT_F; ++c) o[c] = x[c] - lse;
}

// ---------------------------------------------------------------------------
extern "C" void kernel_launch(void* const* d_in, const int* in_sizes, int n_in,
                              void* d_out, int out_size, void* d_ws, size_t ws_size,
                              hipStream_t stream) {
  (void)in_sizes; (void)n_in; (void)out_size; (void)ws_size;
  const float* X   = (const float*)d_in[0];
  const int*   src = (const int*)  d_in[1];
  // d_in[2] = dst (structure known: repeat(arange(N), DEG)) - unused
  const float* W1  = (const float*)d_in[3];
  const float* al1 = (const float*)d_in[4];
  const float* ar1 = (const float*)d_in[5];
  const float* b1  = (const float*)d_in[6];
  const float* Wh  = (const float*)d_in[7];
  const float* alh = (const float*)d_in[8];
  const float* arh = (const float*)d_in[9];
  const float* bh  = (const float*)d_in[10];
  const float* W2  = (const float*)d_in[11];
  const float* al2 = (const float*)d_in[12];
  const float* ar2 = (const float*)d_in[13];
  const float* b2  = (const float*)d_in[14];

  // -------- workspace carve-up (~61 MB), 256B aligned slabs
  char* p = (char*)d_ws;
  auto take = [&](size_t bytes) -> char* {
    char* r = p;
    p += (bytes + 255) & ~(size_t)255;
    return r;
  };
  h16*   hbuf  = (h16*)  take((size_t)NNODES * IN_F * sizeof(h16));    // f16 activations (X, h1, h2)
  float* feat  = (float*)take((size_t)NNODES * HH_F * sizeof(float));  // GEMM output per layer
  float* el    = (float*)take((size_t)NNODES * 4 * sizeof(float));
  float* er    = (float*)take((size_t)NNODES * 4 * sizeof(float));
  float* logit = (float*)take((size_t)NNODES * OUT_F * sizeof(float));
  h16*   W1t   = (h16*)  take((size_t)HH_F   * IN_F * sizeof(h16));
  h16*   Wht   = (h16*)  take((size_t)HH_F   * HH_F * sizeof(h16));
  h16*   W2t   = (h16*)  take((size_t)OUT_PAD* HH_F * sizeof(h16));

  const int MT = NNODES / 16;  // 3125 row panels, exact

  // -------- prep: f16 X, transposed f16 weights
  cvt_f32_to_f16_kernel<<<(NNODES * IN_F + 255) / 256, 256, 0, stream>>>(X, hbuf, NNODES * IN_F);
  transpose_f16_kernel<<<(IN_F * HH_F   + 255) / 256, 256, 0, stream>>>(W1, W1t, IN_F, HH_F, HH_F);
  transpose_f16_kernel<<<(HH_F * HH_F   + 255) / 256, 256, 0, stream>>>(Wh, Wht, HH_F, HH_F, HH_F);
  transpose_f16_kernel<<<(HH_F * OUT_PAD+ 255) / 256, 256, 0, stream>>>(W2, W2t, HH_F, OUT_F, OUT_PAD);

  // -------- layer 1: GATConv(256 -> 32, 4 heads) + ReLU
  wmma_gemm_kernel<IN_F, HH_F / 16><<<MT, 32, 0, stream>>>(hbuf, W1t, feat);
  attn_scores_kernel<<<(NNODES * 4 + 255) / 256, 256, 0, stream>>>(feat, al1, ar1, el, er, 4, 32, HH_F);
  gat_aggregate_kernel<4, 32, HH_F, true><<<NNODES, 128, 0, stream>>>(
      feat, el, er, src, b1, nullptr, hbuf);          // h1 -> f16 in-place slab

  // -------- layer 2: GATConv(128 -> 128, 1 head) + ReLU
  wmma_gemm_kernel<HH_F, HH_F / 16><<<MT, 32, 0, stream>>>(hbuf, Wht, feat);
  attn_scores_kernel<<<(NNODES + 255) / 256, 256, 0, stream>>>(feat, alh, arh, el, er, 1, HH_F, HH_F);
  gat_aggregate_kernel<1, HH_F, HH_F, true><<<NNODES, 128, 0, stream>>>(
      feat, el, er, src, bh, nullptr, hbuf);          // h2 -> f16

  // -------- layer 3: GATConv(128 -> 40, 1 head), cols padded to 48
  wmma_gemm_kernel<HH_F, OUT_PAD / 16><<<MT, 32, 0, stream>>>(hbuf, W2t, feat);
  attn_scores_kernel<<<(NNODES + 255) / 256, 256, 0, stream>>>(feat, al2, ar2, el, er, 1, OUT_F, OUT_PAD);
  gat_aggregate_kernel<1, OUT_F, OUT_PAD, false><<<NNODES, 64, 0, stream>>>(
      feat, el, er, src, b2, logit, nullptr);

  // -------- log_softmax -> d_out
  log_softmax_kernel<<<(NNODES + 255) / 256, 256, 0, stream>>>(logit, (float*)d_out);
}
